// SceneGraphGenerator_49289044689195
// MI455X (gfx1250) — matrix operand order repaired and checked by reading, not compiled, not run on hardware
//
#include <hip/hip_runtime.h>
#include <hip/hip_bf16.h>
#include <math.h>

typedef _Float16 f16;
typedef __attribute__((ext_vector_type(16))) _Float16 v16h;
typedef __attribute__((ext_vector_type(8)))  float    v8f;
typedef __attribute__((ext_vector_type(4)))  int      v4i;

#ifndef USE_ASYNC_LDS
#define USE_ASYNC_LDS 1
#endif

#if USE_ASYNC_LDS
/* gfx1250 async global->LDS copy, 16B per lane, tracked by ASYNCcnt.    */
/* VDST operand = wave-relative LDS byte address (low 32 bits of generic */
/* shared pointer), VADDR = 64-bit global address.                       */
#define ASYNC_CP16(ldsoff, gptr)                                          \
    asm volatile("global_load_async_to_lds_b128 %0, %1, off"              \
                 :: "v"(ldsoff), "v"(gptr) : "memory")
#define ASYNC_WAIT() asm volatile("s_wait_asynccnt 0x0" ::: "memory")
#else
#define ASYNC_WAIT()
#endif

#define B_    8
#define C_    1024
#define H_    64
#define W_    64
#define N_    20
#define P_    7
#define HID_  768
#define NE_   237
#define NR_   310
#define NREL_ 10
#define ROI_DIM (C_*P_*P_)          /* 50176 */
#define NROI  (B_*N_)               /* 160 */
#define NPAIR (B_*N_*N_)            /* 3200 */
#define RELK  (2*HID_ + HID_/4)     /* 1728 */

/* d_out float offsets, in reference return order */
#define O_BOXES  0
#define O_ENT    (O_BOXES + B_*N_*4)
#define O_REGC   (O_ENT   + B_*N_*NE_)
#define O_POS    (O_REGC  + B_*N_*NR_)
#define O_REL    (O_POS   + B_*N_*2)
#define O_SCORES (O_REL   + B_*N_*N_*NREL_)
#define O_CLS    (O_SCORES+ B_*N_)
#define O_REG    (O_CLS   + B_*(NE_+1)*H_*W_)
#define O_CTR    (O_REG   + B_*4*H_*W_)

__device__ __forceinline__ float sigm(float x){ return 1.f/(1.f+__expf(-x)); }

/* ------------------------------------------------------------------ */
/* Generic f16 WMMA GEMM: out[M,N] = A[M,K](f16) * B[K,N](f32) + bias */
/* block 256 = 8 waves; macro tile 160(M) x 64(N); K multiple of 32.  */
/* Double-buffered LDS pipeline: issue(next) -> compute(cur) ->       */
/* commit(next) -> s_wait_asynccnt -> one barrier per K-step.         */
/* ------------------------------------------------------------------ */
__global__ __launch_bounds__(256)
void gemm_wmma_kernel(const f16* __restrict__ A, const float* __restrict__ Bw,
                      const float* __restrict__ bias, void* __restrict__ outp,
                      int outF16, int doRelu, int M, int N, int K)
{
    __shared__ __align__(16) f16 Asub[2][160][32];
    __shared__ __align__(16) f16 Bsub[2][64][32];   /* transposed: [n][k] */
    const int t = threadIdx.x, lane = t & 31, w = t >> 5;
    const int n0 = blockIdx.x * 64;
    const int m0 = blockIdx.y * 160;
    const int nf = w & 3, par = w >> 2;
    const bool fullM = (m0 + 160 <= M);
    const bool fastN = (n0 + 64 <= N) && ((N & 3) == 0);

    /* per-thread staging coordinates (hoisted out of the K loop) */
    const int arow = t >> 2, apart = (t & 3) * 8;        /* A 16B chunks */
    const int bk = t >> 4, bn4 = (t & 15) << 2;          /* B float4 chunks */
#if USE_ASYNC_LDS
    const unsigned aLds0 = (unsigned)(size_t)&Asub[0][arow][apart];
    const unsigned aLds1 = (unsigned)(size_t)&Asub[1][arow][apart];
#endif

    /* guarded (edge-tile) staging helpers */
    auto stageA_slow = [&](int kc, int buf) {
        for (int idx = t; idx < 640; idx += 256) {
            int row = idx >> 2, part = idx & 3;
            int gm = m0 + row;
            v4i val = {0,0,0,0};
            if (gm < M)
                val = *(const v4i*)(A + (size_t)gm * K + (kc << 5) + part * 8);
            *(v4i*)(&Asub[buf][row][part * 8]) = val;
        }
    };
    auto stageB_slow = [&](int kc, int buf) {
        for (int idx = t; idx < 2048; idx += 256) {
            int n = idx & 63, k = idx >> 6;
            float v = 0.f;
            if (n0 + n < N) v = Bw[(size_t)((kc << 5) + k) * N + n0 + n];
            Bsub[buf][n][k] = (f16)v;
        }
    };

    v8f acc[5];
#pragma unroll
    for (int f = 0; f < 5; ++f)
#pragma unroll
        for (int e = 0; e < 8; ++e) acc[f][e] = 0.f;

    const int kT = K >> 5;

    /* ---------------- prologue: stage tile 0 into buffer 0 -------- */
    if (fullM) {
        const f16* g0 = A + (size_t)(m0 + arow) * K + apart;
#if USE_ASYNC_LDS
        ASYNC_CP16(aLds0, g0);
        ASYNC_CP16(aLds0 + 10240u, g0 + (size_t)64 * K);
        if (t < 128) ASYNC_CP16(aLds0 + 20480u, g0 + (size_t)128 * K);
#else
        *(v4i*)&Asub[0][arow][apart]       = *(const v4i*)g0;
        *(v4i*)&Asub[0][arow + 64][apart]  = *(const v4i*)(g0 + (size_t)64 * K);
        if (t < 128)
            *(v4i*)&Asub[0][arow + 128][apart] = *(const v4i*)(g0 + (size_t)128 * K);
#endif
    } else stageA_slow(0, 0);
    if (fastN) {
        const float* gb = Bw + (size_t)bk * N + n0 + bn4;
        float4 v0 = *(const float4*)gb;
        float4 v1 = *(const float4*)(gb + (size_t)16 * N);
        Bsub[0][bn4 + 0][bk] = (f16)v0.x;  Bsub[0][bn4 + 1][bk] = (f16)v0.y;
        Bsub[0][bn4 + 2][bk] = (f16)v0.z;  Bsub[0][bn4 + 3][bk] = (f16)v0.w;
        Bsub[0][bn4 + 0][bk + 16] = (f16)v1.x;  Bsub[0][bn4 + 1][bk + 16] = (f16)v1.y;
        Bsub[0][bn4 + 2][bk + 16] = (f16)v1.z;  Bsub[0][bn4 + 3][bk + 16] = (f16)v1.w;
    } else stageB_slow(0, 0);
    ASYNC_WAIT();
    __syncthreads();

    /* ---------------- pipelined main loop -------------------------- */
    for (int kc = 0; kc < kT; ++kc) {
        const int cur = kc & 1, nxt = cur ^ 1;
        const bool hasNext = (kc + 1) < kT;
        float4 vb0 = {0,0,0,0}, vb1 = {0,0,0,0};
        v4i va0 = {0,0,0,0}, va1 = {0,0,0,0}, va2 = {0,0,0,0};

        /* issue next tile's global traffic (no waits yet) */
        if (hasNext) {
            const int kco = (kc + 1) << 5;
            if (fullM) {
                const f16* g0 = A + (size_t)(m0 + arow) * K + kco + apart;
#if USE_ASYNC_LDS
                unsigned l0 = nxt ? aLds1 : aLds0;
                ASYNC_CP16(l0, g0);
                ASYNC_CP16(l0 + 10240u, g0 + (size_t)64 * K);
                if (t < 128) ASYNC_CP16(l0 + 20480u, g0 + (size_t)128 * K);
#else
                va0 = *(const v4i*)g0;
                va1 = *(const v4i*)(g0 + (size_t)64 * K);
                if (t < 128) va2 = *(const v4i*)(g0 + (size_t)128 * K);
#endif
            }
            if (fastN) {
                const float* gb = Bw + (size_t)(kco + bk) * N + n0 + bn4;
                vb0 = *(const float4*)gb;
                vb1 = *(const float4*)(gb + (size_t)16 * N);
                if (kc + 2 < kT) __builtin_prefetch(gb + (size_t)32 * N, 0, 0);
            }
        }

        /* compute from buffer `cur` (overlaps in-flight loads) */
        {
            v16h bf;
            const f16* bp = &Bsub[cur][nf * 16 + (lane & 15)][(lane >> 4) * 16];
            *(v4i*)&bf       = *(const v4i*)bp;
            *((v4i*)&bf + 1) = *(const v4i*)(bp + 8);
#pragma unroll
            for (int f = 0; f < 5; ++f) {
                int mf = par + 2 * f;
                const f16* ap = &Asub[cur][mf * 16 + (lane & 15)][(lane >> 4) * 16];
                v16h af;
                *(v4i*)&af       = *(const v4i*)ap;
                *((v4i*)&af + 1) = *(const v4i*)(ap + 8);
                acc[f] = __builtin_amdgcn_wmma_f32_16x16x32_f16(
                            false, af, false, bf, (short)0, acc[f], false, false);
            }
        }

        /* commit next tile into LDS */
        if (hasNext) {
            if (!fullM) stageA_slow(kc + 1, nxt);
#if !USE_ASYNC_LDS
            else {
                *(v4i*)&Asub[nxt][arow][apart]      = va0;
                *(v4i*)&Asub[nxt][arow + 64][apart] = va1;
                if (t < 128) *(v4i*)&Asub[nxt][arow + 128][apart] = va2;
            }
#endif
            if (fastN) {
                Bsub[nxt][bn4 + 0][bk] = (f16)vb0.x;  Bsub[nxt][bn4 + 1][bk] = (f16)vb0.y;
                Bsub[nxt][bn4 + 2][bk] = (f16)vb0.z;  Bsub[nxt][bn4 + 3][bk] = (f16)vb0.w;
                Bsub[nxt][bn4 + 0][bk + 16] = (f16)vb1.x;  Bsub[nxt][bn4 + 1][bk + 16] = (f16)vb1.y;
                Bsub[nxt][bn4 + 2][bk + 16] = (f16)vb1.z;  Bsub[nxt][bn4 + 3][bk + 16] = (f16)vb1.w;
            } else stageB_slow(kc + 1, nxt);
        }
        ASYNC_WAIT();
        __syncthreads();
    }

    /* ---------------- epilogue ------------------------------------- */
#pragma unroll
    for (int f = 0; f < 5; ++f) {
        int mf = par + 2 * f;
#pragma unroll
        for (int r = 0; r < 8; ++r) {
            int gm = m0 + mf * 16 + r + ((lane >> 4) << 3);
            int gn = n0 + nf * 16 + (lane & 15);
            if (gm < M && gn < N) {
                float v = acc[f][r] + (bias ? bias[gn] : 0.f);
                if (doRelu && v < 0.f) v = 0.f;
                if (outF16) ((f16*)outp)[(size_t)gm * N + gn] = (f16)v;
                else        ((float*)outp)[(size_t)gm * N + gn] = v;
            }
        }
    }
}

/* ------------------------------------------------------------------ */
/* Implicit-GEMM 3x3 conv, pad=1, NHWC f16, OC=256, k=(dy*3+dx)*Cin+c */
/* grid.x = B*64*2 (b,y,x-half), grid.y = 2 (oc block of 128)         */
/* Double-buffered LDS pipeline like the GEMM kernel.                 */
/* ------------------------------------------------------------------ */
__global__ __launch_bounds__(256)
void conv3x3_wmma_kernel(const f16* __restrict__ src, int Cin,
                         const f16* __restrict__ Aw, const float* __restrict__ bias,
                         f16* __restrict__ dst)
{
    __shared__ __align__(16) f16 Atile[2][128][32];
    __shared__ __align__(16) f16 Btile[2][32][32];  /* [n(spatial)][k] */
    const int t = threadIdx.x, lane = t & 31, w = t >> 5;
    const int bx = blockIdx.x;
    const int b = bx >> 7, y = (bx >> 1) & 63, x0 = (bx & 1) * 32;
    const int oc0 = blockIdx.y * 128;
    const int K = 9 * Cin, cpos = Cin >> 5;
    const int nf = w & 1, of = w >> 1;
    const int arow = t >> 2, apart = (t & 3) * 8;
    const int bn = t >> 3, bcp = (t & 7) << 2;
#if USE_ASYNC_LDS
    const unsigned aLds0 = (unsigned)(size_t)&Atile[0][arow][apart];
    const unsigned aLds1 = (unsigned)(size_t)&Atile[1][arow][apart];
#endif

    /* im2col 8B fetch for this thread's (n, 4-channel) slot of tile kc */
    auto im2col8 = [&](int kc) -> unsigned long long {
        int pos = kc / cpos;
        int c0  = (kc - pos * cpos) << 5;
        int dy  = pos / 3 - 1, dx = pos % 3 - 1;
        int yy = y + dy, xx = x0 + bn + dx;
        unsigned long long v = 0ull;
        if (yy >= 0 && yy < 64 && xx >= 0 && xx < 64)
            v = *(const unsigned long long*)
                (src + (size_t)((b * 64 + yy) * 64 + xx) * Cin + c0 + bcp);
        return v;
    };

    v8f acc[2];
#pragma unroll
    for (int f = 0; f < 2; ++f)
#pragma unroll
        for (int e = 0; e < 8; ++e) acc[f][e] = 0.f;

    const int kT = K >> 5;

    /* prologue: stage tile 0 into buffer 0 */
    {
        const f16* g0 = Aw + (size_t)(oc0 + arow) * K + apart;
#if USE_ASYNC_LDS
        ASYNC_CP16(aLds0, g0);
        ASYNC_CP16(aLds0 + 8192u, g0 + (size_t)64 * K);
#else
        *(v4i*)&Atile[0][arow][apart]      = *(const v4i*)g0;
        *(v4i*)&Atile[0][arow + 64][apart] = *(const v4i*)(g0 + (size_t)64 * K);
#endif
        *(unsigned long long*)(&Btile[0][bn][bcp]) = im2col8(0);
    }
    ASYNC_WAIT();
    __syncthreads();

    for (int kc = 0; kc < kT; ++kc) {
        const int cur = kc & 1, nxt = cur ^ 1;
        const bool hasNext = (kc + 1) < kT;
        unsigned long long vb = 0ull;
        v4i va0 = {0,0,0,0}, va1 = {0,0,0,0};

        /* issue next tile */
        if (hasNext) {
            const f16* g0 = Aw + (size_t)(oc0 + arow) * K + ((kc + 1) << 5) + apart;
#if USE_ASYNC_LDS
            unsigned l0 = nxt ? aLds1 : aLds0;
            ASYNC_CP16(l0, g0);
            ASYNC_CP16(l0 + 8192u, g0 + (size_t)64 * K);
#else
            va0 = *(const v4i*)g0;
            va1 = *(const v4i*)(g0 + (size_t)64 * K);
#endif
            vb = im2col8(kc + 1);
        }

        /* compute from buffer `cur` */
        {
            v16h bf;
            const f16* bp = &Btile[cur][nf * 16 + (lane & 15)][(lane >> 4) * 16];
            *(v4i*)&bf       = *(const v4i*)bp;
            *((v4i*)&bf + 1) = *(const v4i*)(bp + 8);
#pragma unroll
            for (int f = 0; f < 2; ++f) {
                int mf = 2 * of + f;
                const f16* ap = &Atile[cur][mf * 16 + (lane & 15)][(lane >> 4) * 16];
                v16h af;
                *(v4i*)&af       = *(const v4i*)ap;
                *((v4i*)&af + 1) = *(const v4i*)(ap + 8);
                acc[f] = __builtin_amdgcn_wmma_f32_16x16x32_f16(
                            false, af, false, bf, (short)0, acc[f], false, false);
            }
        }

        /* commit next tile */
        if (hasNext) {
#if !USE_ASYNC_LDS
            *(v4i*)&Atile[nxt][arow][apart]      = va0;
            *(v4i*)&Atile[nxt][arow + 64][apart] = va1;
#endif
            *(unsigned long long*)(&Btile[nxt][bn][bcp]) = vb;
        }
        ASYNC_WAIT();
        __syncthreads();
    }

#pragma unroll
    for (int f = 0; f < 2; ++f)
#pragma unroll
        for (int r = 0; r < 8; ++r) {
            int oc = oc0 + (2 * of + f) * 16 + r + ((lane >> 4) << 3);
            int x  = x0 + nf * 16 + (lane & 15);
            float v = acc[f][r] + bias[oc];
            if (v < 0.f) v = 0.f;
            dst[(size_t)((b * 64 + y) * 64 + x) * 256 + oc] = (f16)v;
        }
}

/* ----------------------- setup / scalar kernels -------------------- */

__global__ void pack_conv_w(const float* __restrict__ wsrc, const float* __restrict__ g,
                            const float* __restrict__ vv, f16* __restrict__ outw, int Cin)
{
    int idx = blockIdx.x * 256 + threadIdx.x;
    int Kc = 9 * Cin;
    if (idx >= 256 * Kc) return;
    int oc = idx / Kc, r = idx % Kc;
    int pos = r / Cin, c = r % Cin;              /* k = pos*Cin + c */
    float s = g[oc] * rsqrtf(vv[oc] + 1e-5f);
    outw[(size_t)oc * Kc + r] = (f16)(wsrc[((size_t)oc * Cin + c) * 9 + pos] * s);
}

__global__ void make_bias(const float* cb, const float* g, const float* bb,
                          const float* m, const float* vv, float* outb)
{
    int oc = threadIdx.x;
    float s = g[oc] * rsqrtf(vv[oc] + 1e-5f);
    outb[oc] = cb[oc] * s + bb[oc] - m[oc] * s;
}

__global__ void pack_heads(const float* clsw, const float* regw, const float* ctrw,
                           float* Wh)
{
    int idx = blockIdx.x * 256 + threadIdx.x;    /* 256(c) x 256(n) */
    int c = idx >> 8, n = idx & 255;
    float v = 0.f;
    if (n < NE_ + 1)      v = clsw[n * 256 + c];
    else if (n < NE_ + 5) v = regw[(n - (NE_ + 1)) * 256 + c];
    else if (n == NE_ + 5) v = ctrw[c];
    Wh[c * 256 + n] = v;                         /* [K=c][N=n] */
}

__global__ void pack_head_bias(const float* cb, const float* rb, const float* tb,
                               float* biash)
{
    int n = threadIdx.x;
    float v = 0.f;
    if (n < NE_ + 1) v = cb[n];
    else if (n < NE_ + 5) v = rb[n - (NE_ + 1)];
    else if (n == NE_ + 5) v = tb[0];
    biash[n] = v;
}

__global__ void vf_to_nhwc(const float* __restrict__ vf, f16* __restrict__ out)
{
    size_t d = (size_t)blockIdx.x * 256 + threadIdx.x;   /* B*H*W*C */
    int c = d & 1023;
    size_t rest = d >> 10;
    int x = rest & 63, y = (rest >> 6) & 63, b = rest >> 12;
    out[d] = (f16)vf[(((size_t)b * 1024 + c) * 64 + y) * 64 + x];
}

__global__ void head_scatter(const float* __restrict__ hout, float* __restrict__ out)
{
    size_t idx = (size_t)blockIdx.x * 256 + threadIdx.x; /* B*4096*256 */
    int n = idx & 255;
    size_t sp = idx >> 8;
    int b = sp >> 12, yx = sp & 4095;
    float v = hout[idx];
    if (n < NE_ + 1)
        out[O_CLS + ((size_t)(b * (NE_ + 1) + n)) * 4096 + yx] = v;
    else if (n < NE_ + 5)
        out[O_REG + ((size_t)(b * 4 + (n - (NE_ + 1)))) * 4096 + yx] = v;
    else if (n == NE_ + 5)
        out[O_CTR + (size_t)b * 4096 + yx] = v;
}

__global__ void integral_rows(const float* __restrict__ vf, float* __restrict__ S)
{
    int idx = blockIdx.x * 256 + threadIdx.x;    /* B*C*65 */
    if (idx >= B_ * C_ * 65) return;
    int bc = idx / 65, yy = idx % 65;
    float* row = S + ((size_t)bc * 65 + yy) * 65;
    if (yy == 0) { for (int x = 0; x < 65; ++x) row[x] = 0.f; return; }
    const float* srow = vf + (size_t)bc * 4096 + (size_t)(yy - 1) * 64;
    float a = 0.f; row[0] = 0.f;
    for (int x = 0; x < 64; ++x) { a += srow[x]; row[x + 1] = a; }
}

__global__ void integral_cols(float* __restrict__ S)
{
    int idx = blockIdx.x * 256 + threadIdx.x;    /* B*C*65 */
    if (idx >= B_ * C_ * 65) return;
    int bc = idx / 65, xx = idx % 65;
    float a = 0.f;
    for (int yy = 0; yy < 65; ++yy) {
        size_t o = ((size_t)bc * 65 + yy) * 65 + xx;
        a += S[o]; S[o] = a;
    }
}

__global__ __launch_bounds__(256)
void topk_kernel(float* __restrict__ o, float* __restrict__ boxes_ws,
                 int* __restrict__ bins)
{
    __shared__ float sv[4096];
    __shared__ float rv[256];
    __shared__ int   ri[256];
    const int b = blockIdx.x, t = threadIdx.x;
    for (int i = t; i < 4096; i += 256) sv[i] = o[O_CTR + b * 4096 + i];
    __syncthreads();
    for (int k = 0; k < N_; ++k) {
        float bv = -3.4e38f; int bi = 0x7fffffff;
        for (int i = t; i < 4096; i += 256) {
            float v = sv[i];
            if (v > bv) { bv = v; bi = i; }
        }
        rv[t] = bv; ri[t] = bi;
        __syncthreads();
        for (int s = 128; s > 0; s >>= 1) {
            if (t < s) {
                if (rv[t + s] > rv[t] || (rv[t + s] == rv[t] && ri[t + s] < ri[t])) {
                    rv[t] = rv[t + s]; ri[t] = ri[t + s];
                }
            }
            __syncthreads();
        }
        if (t == 0) {
            int idx = ri[0]; float raw = rv[0];
            o[O_SCORES + b * N_ + k] = sigm(raw);
            float bb[4];
            for (int cmp = 0; cmp < 4; ++cmp) {
                float rr = o[O_REG + (size_t)(b * 4 + cmp) * 4096 + idx];
                bb[cmp] = sigm(rr);
                o[O_BOXES + (b * N_ + k) * 4 + cmp] = bb[cmp];
                boxes_ws[(b * N_ + k) * 4 + cmp] = bb[cmp];
            }
            int x1 = min(max((int)floorf(bb[0] * 64.f), 0), 63);
            int y1 = min(max((int)floorf(bb[1] * 64.f), 0), 63);
            int x2 = max(x1 + 1, min((int)floorf(bb[2] * 64.f) + 1, 64));
            int y2 = max(y1 + 1, min((int)floorf(bb[3] * 64.f) + 1, 64));
            int iw = x2 - x1, ih = y2 - y1;
            int* bp = bins + (b * N_ + k) * 28;
            for (int i = 0; i < 7; ++i) {
                bp[i]      = x1 + (i * iw) / 7;
                bp[7 + i]  = x1 + ((i + 1) * iw + 6) / 7;
                bp[14 + i] = y1 + (i * ih) / 7;
                bp[21 + i] = y1 + ((i + 1) * ih + 6) / 7;
            }
            sv[idx] = -3.4e38f;
        }
        __syncthreads();
    }
}

__global__ void build_roi_kernel(const float* __restrict__ S, const int* __restrict__ bins,
                                 f16* __restrict__ roi)
{
    size_t idx = (size_t)blockIdx.x * 256 + threadIdx.x;  /* 160*50176 */
    int r = idx % ROI_DIM;
    int row = idx / ROI_DIM;
    int c = r / 49, q = r % 49, i = q / 7, j = q % 7;
    int b = row / N_;
    const int* bp = bins + row * 28;
    int xs = bp[j], xe = bp[7 + j], ys = bp[14 + i], ye = bp[21 + i];
    const float* Sp = S + ((size_t)(b * 1024 + c)) * 65 * 65;
    float s = Sp[ye * 65 + xe] - Sp[ys * 65 + xe] - Sp[ye * 65 + xs] + Sp[ys * 65 + xs];
    float area = (float)((ye - ys) * (xe - xs));
    roi[idx] = (f16)(s / area);
}

__global__ void sp1_kernel(const float* __restrict__ boxes_ws, const float* __restrict__ w1,
                           const float* __restrict__ b1, f16* __restrict__ sph)
{
    int idx = blockIdx.x * 256 + threadIdx.x;    /* 3200*64 */
    int o = idx & 63, row = idx >> 6;
    int b = row / (N_ * N_), rr = row % (N_ * N_), n1 = rr / N_, n2 = rr % N_;
    const float* bs = boxes_ws + (b * N_ + n1) * 4;
    const float* bo = boxes_ws + (b * N_ + n2) * 4;
    float a = b1[o];
    for (int q = 0; q < 4; ++q) a += bs[q] * w1[q * 64 + o];
    for (int q = 0; q < 4; ++q) a += bo[q] * w1[(4 + q) * 64 + o];
    sph[idx] = (f16)(a > 0.f ? a : 0.f);
}

__global__ void build_relin_kernel(const f16* __restrict__ obj, const f16* __restrict__ spo,
                                   f16* __restrict__ relin)
{
    size_t idx = (size_t)blockIdx.x * 256 + threadIdx.x;  /* 3200*1728 */
    int k = idx % RELK;
    int row = idx / RELK;
    int b = row / (N_ * N_), rr = row % (N_ * N_), n1 = rr / N_, n2 = rr % N_;
    f16 v;
    if (k < HID_)          v = obj[(size_t)(b * N_ + n1) * HID_ + k];
    else if (k < 2 * HID_) v = obj[(size_t)(b * N_ + n2) * HID_ + (k - HID_)];
    else                   v = spo[(size_t)row * (HID_ / 4) + (k - 2 * HID_)];
    relin[idx] = v;
}

/* ------------------------------- launcher -------------------------- */
extern "C" void kernel_launch(void* const* d_in, const int* in_sizes, int n_in,
                              void* d_out, int out_size, void* d_ws, size_t ws_size,
                              hipStream_t stream)
{
    const float* vf      = (const float*)d_in[0];
    const float* conv1_w = (const float*)d_in[1];
    const float* conv1_b = (const float*)d_in[2];
    const float* bn1_g   = (const float*)d_in[3];
    const float* bn1_b   = (const float*)d_in[4];
    const float* bn1_m   = (const float*)d_in[5];
    const float* bn1_v   = (const float*)d_in[6];
    const float* conv2_w = (const float*)d_in[7];
    const float* conv2_b = (const float*)d_in[8];
    const float* bn2_g   = (const float*)d_in[9];
    const float* bn2_b   = (const float*)d_in[10];
    const float* bn2_m   = (const float*)d_in[11];
    const float* bn2_v   = (const float*)d_in[12];
    const float* cls_w   = (const float*)d_in[13];
    const float* cls_b   = (const float*)d_in[14];
    const float* reg_w   = (const float*)d_in[15];
    const float* reg_b   = (const float*)d_in[16];
    const float* ctr_w   = (const float*)d_in[17];
    const float* ctr_b   = (const float*)d_in[18];
    const float* ent_w1  = (const float*)d_in[19];
    const float* ent_b1  = (const float*)d_in[20];
    const float* ent_w2  = (const float*)d_in[21];
    const float* ent_b2  = (const float*)d_in[22];
    const float* regc_w1 = (const float*)d_in[23];
    const float* regc_b1 = (const float*)d_in[24];
    const float* regc_w2 = (const float*)d_in[25];
    const float* regc_b2 = (const float*)d_in[26];
    const float* pos_w1  = (const float*)d_in[27];
    const float* pos_b1  = (const float*)d_in[28];
    const float* pos_w2  = (const float*)d_in[29];
    const float* pos_b2  = (const float*)d_in[30];
    const float* objp_w  = (const float*)d_in[31];
    const float* objp_b  = (const float*)d_in[32];
    const float* rel_w1  = (const float*)d_in[33];
    const float* rel_b1  = (const float*)d_in[34];
    const float* rel_w2  = (const float*)d_in[35];
    const float* rel_b2  = (const float*)d_in[36];
    const float* sp_w1   = (const float*)d_in[37];
    const float* sp_b1   = (const float*)d_in[38];
    const float* sp_w2   = (const float*)d_in[39];
    const float* sp_b2   = (const float*)d_in[40];

    float* out = (float*)d_out;

    /* workspace carve (256B aligned) */
    char* p = (char*)d_ws;
    auto alloc = [&](size_t bytes) -> void* {
        void* r = (void*)p;
        p += (bytes + 255) & ~(size_t)255;
        return r;
    };
    f16*   vf16   = (f16*)alloc((size_t)B_*H_*W_*C_*2);
    f16*   A1     = (f16*)alloc((size_t)256*9*C_*2);
    float* bias1  = (float*)alloc(256*4);
    f16*   A2     = (f16*)alloc((size_t)256*9*256*2);
    float* bias2  = (float*)alloc(256*4);
    float* Wh     = (float*)alloc((size_t)256*256*4);
    float* biash  = (float*)alloc(256*4);
    f16*   r1     = (f16*)alloc((size_t)B_*H_*W_*256*2);
    f16*   r2     = (f16*)alloc((size_t)B_*H_*W_*256*2);
    float* hout   = (float*)alloc((size_t)B_*H_*W_*256*4);
    float* S      = (float*)alloc((size_t)B_*C_*65*65*4);
    float* boxesw = (float*)alloc((size_t)NROI*4*4);
    int*   bins   = (int*)alloc((size_t)NROI*28*4);
    f16*   roi    = (f16*)alloc((size_t)NROI*ROI_DIM*2);
    f16*   ent_h  = (f16*)alloc((size_t)NROI*HID_*2);
    f16*   regc_h = (f16*)alloc((size_t)NROI*HID_*2);
    f16*   pos_h  = (f16*)alloc((size_t)NROI*256*2);
    f16*   objh   = (f16*)alloc((size_t)NROI*HID_*2);
    f16*   sph    = (f16*)alloc((size_t)NPAIR*64*2);
    f16*   spo    = (f16*)alloc((size_t)NPAIR*(HID_/4)*2);
    f16*   relin  = (f16*)alloc((size_t)NPAIR*RELK*2);
    f16*   relh   = (f16*)alloc((size_t)NPAIR*HID_*2);

    const dim3 blk(256);

    /* weight packing + BN folding */
    pack_conv_w<<<(256*9*C_+255)/256, blk, 0, stream>>>(conv1_w, bn1_g, bn1_v, A1, C_);
    make_bias<<<1, blk, 0, stream>>>(conv1_b, bn1_g, bn1_b, bn1_m, bn1_v, bias1);
    pack_conv_w<<<(256*9*256+255)/256, blk, 0, stream>>>(conv2_w, bn2_g, bn2_v, A2, 256);
    make_bias<<<1, blk, 0, stream>>>(conv2_b, bn2_g, bn2_b, bn2_m, bn2_v, bias2);
    pack_heads<<<256, blk, 0, stream>>>(cls_w, reg_w, ctr_w, Wh);
    pack_head_bias<<<1, blk, 0, stream>>>(cls_b, reg_b, ctr_b, biash);

    /* backbone */
    vf_to_nhwc<<<(B_*H_*W_*C_)/256, blk, 0, stream>>>(vf, vf16);
    conv3x3_wmma_kernel<<<dim3(B_*H_*2, 2), blk, 0, stream>>>(vf16, C_,  A1, bias1, r1);
    conv3x3_wmma_kernel<<<dim3(B_*H_*2, 2), blk, 0, stream>>>(r1,   256, A2, bias2, r2);

    /* 1x1 heads as GEMM, then scatter to NCHW outputs */
    gemm_wmma_kernel<<<dim3(4, (B_*H_*W_+159)/160), blk, 0, stream>>>(
        r2, Wh, biash, hout, 0, 0, B_*H_*W_, 256, 256);
    head_scatter<<<(B_*H_*W_*256)/256, blk, 0, stream>>>(hout, out);

    /* integral image + proposal selection + ROI pooling */
    integral_rows<<<(B_*C_*65+255)/256, blk, 0, stream>>>(vf, S);
    integral_cols<<<(B_*C_*65+255)/256, blk, 0, stream>>>(S);
    topk_kernel<<<B_, blk, 0, stream>>>(out, boxesw, bins);
    build_roi_kernel<<<(NROI*ROI_DIM)/256, blk, 0, stream>>>(S, bins, roi);

    /* ROI MLPs, first layers (f16 out) */
    gemm_wmma_kernel<<<dim3(12, 1), blk, 0, stream>>>(roi, ent_w1,  ent_b1,  ent_h,  1, 1, NROI, HID_, ROI_DIM);
    gemm_wmma_kernel<<<dim3(12, 1), blk, 0, stream>>>(roi, regc_w1, regc_b1, regc_h, 1, 1, NROI, HID_, ROI_DIM);
    gemm_wmma_kernel<<<dim3(4,  1), blk, 0, stream>>>(roi, pos_w1,  pos_b1,  pos_h,  1, 1, NROI, 256,  ROI_DIM);
    gemm_wmma_kernel<<<dim3(12, 1), blk, 0, stream>>>(roi, objp_w,  objp_b,  objh,   1, 0, NROI, HID_, ROI_DIM);

    /* second layers (f32 out to d_out) */
    gemm_wmma_kernel<<<dim3((NE_+63)/64, 1), blk, 0, stream>>>(ent_h,  ent_w2,  ent_b2,  out + O_ENT,  0, 0, NROI, NE_, HID_);
    gemm_wmma_kernel<<<dim3((NR_+63)/64, 1), blk, 0, stream>>>(regc_h, regc_w2, regc_b2, out + O_REGC, 0, 0, NROI, NR_, HID_);
    gemm_wmma_kernel<<<dim3(1, 1), blk, 0, stream>>>(pos_h, pos_w2, pos_b2, out + O_POS, 0, 0, NROI, 2, 256);

    /* relation branch */
    sp1_kernel<<<(NPAIR*64)/256, blk, 0, stream>>>(boxesw, sp_w1, sp_b1, sph);
    gemm_wmma_kernel<<<dim3(3, NPAIR/160), blk, 0, stream>>>(sph, sp_w2, sp_b2, spo, 1, 0, NPAIR, HID_/4, 64);
    build_relin_kernel<<<(NPAIR*RELK)/256, blk, 0, stream>>>(objh, spo, relin);
    gemm_wmma_kernel<<<dim3(12, NPAIR/160), blk, 0, stream>>>(relin, rel_w1, rel_b1, relh, 1, 1, NPAIR, HID_, RELK);
    gemm_wmma_kernel<<<dim3(1, NPAIR/160), blk, 0, stream>>>(relh, rel_w2, rel_b2, out + O_REL, 0, 0, NPAIR, NREL_, HID_);
}
